// MultiHeadAttention_76982993813763
// MI455X (gfx1250) — matrix operand order repaired and checked by reading, not compiled, not run on hardware
//
#include <hip/hip_runtime.h>
#include <hip/hip_bf16.h>

// ---------------------------------------------------------------------------
// MI455X / gfx1250 multi-head attention, bf16 WMMA pipeline with
// double-buffered async (ASYNCcnt) LDS staging.
//   B=4, S=2048, D=1024, H=16, HD=64
// ---------------------------------------------------------------------------

typedef __attribute__((ext_vector_type(16))) __bf16 v16bf;
typedef __attribute__((ext_vector_type(8)))  __bf16 v8bf;
typedef __attribute__((ext_vector_type(4)))  __bf16 v4bf;
typedef __attribute__((ext_vector_type(8)))  float  v8f;
typedef __attribute__((ext_vector_type(4)))  int    v4i;

#define WMMA_BF16(a, b, c) \
  __builtin_amdgcn_wmma_f32_16x16x32_bf16(false, (a), false, (b), (short)0, (c), false, false)

// --- feature probes --------------------------------------------------------
#if defined(__has_builtin)
#if __has_builtin(__builtin_amdgcn_global_load_async_to_lds_b128)
#define HAVE_ASYNC_LDS 1
#endif
#if __has_builtin(__builtin_amdgcn_permlane16)
#define HAVE_PERMLANE16 1
#endif
#endif

// --- async (ASYNCcnt-tracked) global -> LDS 16-byte copy -------------------
__device__ inline void lds_copy8(__bf16* l, const __bf16* g) {
#if defined(HAVE_ASYNC_LDS)
  __builtin_amdgcn_global_load_async_to_lds_b128(
      (__attribute__((address_space(1))) v4i*)g,
      (__attribute__((address_space(3))) v4i*)l, 0, 0);
#else
  *(v8bf*)l = *(const v8bf*)g;
#endif
}

// wait until at most N async copies remain in flight (per wave)
template <int N>
__device__ inline void async_wait_le() {
#if defined(HAVE_ASYNC_LDS)
#if __has_builtin(__builtin_amdgcn_s_wait_asynccnt)
  __builtin_amdgcn_s_wait_asynccnt(N);
#else
  asm volatile("s_wait_asynccnt %0" ::"i"(N) : "memory");
#endif
#endif
}

// --- VALU xor-butterfly within 16-lane groups ------------------------------
constexpr unsigned long long permsel_xor(int m) {
  unsigned long long s = 0;
  for (int i = 0; i < 16; ++i)
    s |= (unsigned long long)((i ^ m) & 15) << (4 * i);
  return s;
}

template <int M>
__device__ inline float xshfl16(float v) {
#if defined(HAVE_PERMLANE16)
  constexpr unsigned long long s = permsel_xor(M);
  int r = __builtin_amdgcn_permlane16(
      __builtin_bit_cast(int, v), __builtin_bit_cast(int, v),
      (unsigned)(s & 0xffffffffu), (unsigned)(s >> 32), false, false);
  return __builtin_bit_cast(float, r);
#else
  return __shfl_xor(v, M, 32);
#endif
}

__device__ inline float grp16_max(float v) {
  v = fmaxf(v, xshfl16<1>(v));
  v = fmaxf(v, xshfl16<2>(v));
  v = fmaxf(v, xshfl16<4>(v));
  v = fmaxf(v, xshfl16<8>(v));
  return v;
}
__device__ inline float grp16_sum(float v) {
  v += xshfl16<1>(v);
  v += xshfl16<2>(v);
  v += xshfl16<4>(v);
  v += xshfl16<8>(v);
  return v;
}

// --- WMMA fragment loaders (contiguous k order, same for A and B) ----------------------------
// A 16x32 bf16: lane l: row = l&15, h = l>>4; VGPR0-7 = K 16h..16h+15
// (contiguous, two 16B loads).
__device__ inline v16bf load_a16x32(const __bf16* base, int row, int stride,
                                    int kbase, int h) {
  const __bf16* p = base + row * stride + kbase + 16 * h;
  union { v16bf v; v8bf h2[2]; } u;
  u.h2[0] = *(const v8bf*)p;
  u.h2[1] = *(const v8bf*)(p + 8);
  return u.v;
}
// B 32x16 bf16 from an [N][K]-major tile: lane l: n=l&15, kh=l>>4;
// VGPR j = (B[16kh+2j,n], B[16kh+2j+1,n]) -> 16 contiguous K elems (32B).
__device__ inline v16bf load_b32x16(const __bf16* base, int n, int stride,
                                    int kbase) {
  return *(const v16bf*)(base + n * stride + kbase);
}

// --- f32 -> bf16 (activations) ---------------------------------------------
__global__ __launch_bounds__(256)
void cvt_f32_bf16(const float* __restrict__ in, __bf16* __restrict__ out, int n4) {
  int i = blockIdx.x * blockDim.x + threadIdx.x;
  if (i < n4) {
    float4 f = ((const float4*)in)[i];
    v4bf o;
    o[0] = (__bf16)f.x; o[1] = (__bf16)f.y; o[2] = (__bf16)f.z; o[3] = (__bf16)f.w;
    ((v4bf*)out)[i] = o;
  }
}

// --- f32 [K][N] -> bf16 [N][K] weight transpose (done once per weight) -----
__global__ __launch_bounds__(256)
void cvt_transpose_w(const float* __restrict__ in, __bf16* __restrict__ out) {
  __shared__ float t[32][33];
  const int D = 1024;
  int bn = blockIdx.x * 32, bk = blockIdx.y * 32;
  int tx = threadIdx.x & 31, ty = threadIdx.x >> 5;  // 32 x 8
#pragma unroll
  for (int i = 0; i < 4; ++i)
    t[ty + 8 * i][tx] = in[(size_t)(bk + ty + 8 * i) * D + bn + tx];
  __syncthreads();
#pragma unroll
  for (int i = 0; i < 4; ++i)
    out[(size_t)(bn + ty + 8 * i) * D + bk + tx] = (__bf16)t[tx][ty + 8 * i];
}

// --- tiled bf16 GEMM: C = A[MxK] * Wt[NxK]^T + bias ------------------------
// 128x128 block, K-step 32, 8 waves (4 along M x 2 along N), 8 WMMA/wave/step.
// 2-stage pipeline: async copies for stage i+1 overlap WMMA on stage i.
// MODE 0: bf16 out, head-split  [B,H,S,HD]   (Q, K projections)
// MODE 1: bf16 out, transposed  [B,H,HD,S]   (V projection)
// MODE 2: f32  out, row-major   [M,N]        (final output GEMM)
template <int MODE>
__global__ __launch_bounds__(256)
void gemm_bf16(const __bf16* __restrict__ A, const __bf16* __restrict__ Wt,
               const float* __restrict__ bias, void* __restrict__ Out,
               int M, int N, int K) {
  __shared__ __align__(32) __bf16 sA[2][128 * 32];   // [m][k]
  __shared__ __align__(32) __bf16 sB[2][128 * 32];   // [n][k]

  const int tid  = threadIdx.x;
  const int wid  = tid >> 5;
  const int lane = tid & 31;
  const int ln   = lane & 15, lh = lane >> 4;
  const int m0 = blockIdx.x * 128, n0 = blockIdx.y * 128;
  const int wm = (wid & 3) * 32;   // 4 waves along M: 2 m-tiles
  const int wn = (wid >> 2) * 64;  // 2 waves along N: 4 n-tiles
  const int r = tid >> 1, cc = (tid & 1) * 16;   // cooperative copy slot

  v8f acc[2][4] = {};

  auto issue = [&](int k0, int buf) {
    const __bf16* ga = A  + (size_t)(m0 + r) * K + k0 + cc;
    const __bf16* gb = Wt + (size_t)(n0 + r) * K + k0 + cc;
    lds_copy8(&sA[buf][r * 32 + cc],     ga);
    lds_copy8(&sA[buf][r * 32 + cc + 8], ga + 8);
    lds_copy8(&sB[buf][r * 32 + cc],     gb);
    lds_copy8(&sB[buf][r * 32 + cc + 8], gb + 8);
  };

  const int niter = K / 32;
  issue(0, 0);                       // prologue: stage 0
  for (int it = 0; it < niter; ++it) {
    const int buf = it & 1;
    if (it + 1 < niter) {
      issue((it + 1) * 32, buf ^ 1); // overlap next stage with this compute
      async_wait_le<4>();            // stage it resident (in-order completion)
    } else {
      async_wait_le<0>();
    }
    __syncthreads();

#pragma unroll
    for (int mt = 0; mt < 2; ++mt) {
      v16bf a = load_a16x32(sA[buf], wm + mt * 16 + ln, 32, 0, lh);
#pragma unroll
      for (int nt = 0; nt < 4; ++nt) {
        v16bf b = load_b32x16(sB[buf], wn + nt * 16 + ln, 32, 16 * lh);
        acc[mt][nt] = WMMA_BF16(a, b, acc[mt][nt]);
      }
    }
    __syncthreads();   // all reads of buf done before it is re-targeted
  }

  // epilogue: C layout lane l: col n = l&15, rows m = vgpr + 8*(l>>4)
#pragma unroll
  for (int mt = 0; mt < 2; ++mt)
#pragma unroll
    for (int nt = 0; nt < 4; ++nt) {
      int n_g = n0 + wn + nt * 16 + ln;
      float bv = bias[n_g];
#pragma unroll
      for (int j = 0; j < 8; ++j) {
        int m_g = m0 + wm + mt * 16 + 8 * lh + j;
        float v = acc[mt][nt][j] + bv;
        if (MODE == 2) {
          ((float*)Out)[(size_t)m_g * N + n_g] = v;
        } else {
          int b = m_g >> 11, s = m_g & 2047;   // S = 2048
          int h = n_g >> 6,  hd = n_g & 63;    // HD = 64
          size_t off = (MODE == 0)
              ? ((size_t)((b * 16 + h) * 2048 + s) * 64 + hd)
              : ((size_t)((b * 16 + h) * 64 + hd) * 2048 + s);
          ((__bf16*)Out)[off] = (__bf16)v;
        }
      }
    }
}

// --- flash attention: softmax(Q K^T / 8) V, k-blocks of 64, pipelined ------
// Q,K: [B,H,S,HD]  V: [B,H,HD,S]  Ctx: [B,S,H*HD] bf16
// 8 waves x 16 q-rows = 128 q rows per block; 16 WMMA per wave per k-block.
__global__ __launch_bounds__(256)
void flash_attn(const __bf16* __restrict__ Q, const __bf16* __restrict__ Km,
                const __bf16* __restrict__ Vt, __bf16* __restrict__ Ctx) {
  const int S = 2048, HD = 64, H = 16;
  __shared__ __align__(32) __bf16 sK[2][64 * 64];    // [k][hd]
  __shared__ __align__(32) __bf16 sV[2][64 * 64];    // [hd][k]  (V^T tile)
  __shared__ __align__(32) __bf16 sP[8 * 16 * 64];   // per-wave P staging

  const int tid = threadIdx.x, wid = tid >> 5, lane = tid & 31;
  const int ln = lane & 15, lh = lane >> 4;
  const int b = blockIdx.z, h = blockIdx.y;
  const int q0 = blockIdx.x * 128 + wid * 16;
  const size_t bh = (size_t)(b * H + h);
  const __bf16* Qb = Q  + bh * S * HD;
  const __bf16* Kb = Km + bh * S * HD;
  const __bf16* Vb = Vt + bh * HD * S;
  const int rr = tid >> 2, cc = (tid & 3) * 16;  // cooperative copy slot

  v16bf qf[2];
#pragma unroll
  for (int c = 0; c < 2; ++c)
    qf[c] = load_a16x32(Qb, q0 + ln, HD, 32 * c, lh);

  v8f o[4] = {};
  float mrow[8], lrow[8];
#pragma unroll
  for (int j = 0; j < 8; ++j) { mrow[j] = -1e30f; lrow[j] = 0.f; }
  const float SCL2 = 0.125f * 1.44269504f;  // (1/sqrt(64)) * log2(e)

  auto issue = [&](int k0, int buf) {
    const __bf16* gk = Kb + (size_t)(k0 + rr) * HD + cc;
    const __bf16* gv = Vb + (size_t)rr * S + k0 + cc;
    lds_copy8(&sK[buf][rr * 64 + cc],     gk);
    lds_copy8(&sK[buf][rr * 64 + cc + 8], gk + 8);
    lds_copy8(&sV[buf][rr * 64 + cc],     gv);
    lds_copy8(&sV[buf][rr * 64 + cc + 8], gv + 8);
  };

  const int niter = S / 64;
  issue(0, 0);                        // prologue: stage 0
  for (int it = 0; it < niter; ++it) {
    const int buf = it & 1;
    const int k0 = it * 64;
    if (it + 1 < niter) {
      issue(k0 + 64, buf ^ 1);        // overlap next stage with this compute
      async_wait_le<4>();
    } else {
      async_wait_le<0>();
    }
    if (k0 + 128 < S) {               // L2 prefetch two blocks ahead
      __builtin_prefetch(Kb + (size_t)(k0 + 128 + rr) * HD, 0, 1);
      __builtin_prefetch(Vb + (size_t)rr * S + k0 + 128, 0, 1);
    }
    __syncthreads();

    // S_tile = Q * K^T : 4 column tiles of 16 k each
    v8f st[4] = {};
#pragma unroll
    for (int c = 0; c < 2; ++c)
#pragma unroll
      for (int nt = 0; nt < 4; ++nt) {
        v16bf bf = load_b32x16(sK[buf], nt * 16 + ln, 64, 32 * c + 16 * lh);
        st[nt] = WMMA_BF16(qf[c], bf, st[nt]);
      }

    // online softmax in exp2 domain; row stats replicated per 16-lane group
    __bf16* pw = &sP[wid * 16 * 64];
#pragma unroll
    for (int j = 0; j < 8; ++j) {
      float t0 = st[0][j] * SCL2, t1 = st[1][j] * SCL2;
      float t2 = st[2][j] * SCL2, t3 = st[3][j] * SCL2;
      float mx = grp16_max(fmaxf(fmaxf(t0, t1), fmaxf(t2, t3)));
      float mnew = fmaxf(mrow[j], mx);
      float corr = exp2f(mrow[j] - mnew);
      float e0 = exp2f(t0 - mnew), e1 = exp2f(t1 - mnew);
      float e2 = exp2f(t2 - mnew), e3 = exp2f(t3 - mnew);
      float rs = grp16_sum((e0 + e1) + (e2 + e3));
      lrow[j] = lrow[j] * corr + rs;
      mrow[j] = mnew;
#pragma unroll
      for (int nt = 0; nt < 4; ++nt) o[nt][j] *= corr;
      int rp = 8 * lh + j;  // P staged [16][64] per wave
      pw[rp * 64 + ln]      = (__bf16)e0;
      pw[rp * 64 + 16 + ln] = (__bf16)e1;
      pw[rp * 64 + 32 + ln] = (__bf16)e2;
      pw[rp * 64 + 48 + ln] = (__bf16)e3;
    }
    asm volatile("s_wait_dscnt 0x0" ::: "memory");  // intra-wave LDS RAW
    __builtin_amdgcn_wave_barrier();

    // O += P * V  (two 32-wide K chunks x 4 hd tiles)
#pragma unroll
    for (int kc = 0; kc < 2; ++kc) {
      v16bf pf = load_a16x32(pw, ln, 64, 32 * kc, lh);
#pragma unroll
      for (int nt = 0; nt < 4; ++nt) {
        v16bf vf = load_b32x16(sV[buf], nt * 16 + ln, 64, 32 * kc + 16 * lh);
        o[nt] = WMMA_BF16(pf, vf, o[nt]);
      }
    }
    __syncthreads();   // all reads of buf done before it is re-targeted
  }

  // normalize and write context [B,S,H*HD]
#pragma unroll
  for (int j = 0; j < 8; ++j) lrow[j] = 1.f / lrow[j];
  __bf16* cb = Ctx + (size_t)b * S * (H * HD) + (size_t)h * HD;
#pragma unroll
  for (int nt = 0; nt < 4; ++nt)
#pragma unroll
    for (int j = 0; j < 8; ++j) {
      int q = q0 + 8 * lh + j;
      cb[(size_t)q * (H * HD) + nt * 16 + ln] = (__bf16)(o[nt][j] * lrow[j]);
    }
}

// ---------------------------------------------------------------------------
extern "C" void kernel_launch(void* const* d_in, const int* in_sizes, int n_in,
                              void* d_out, int out_size, void* d_ws, size_t ws_size,
                              hipStream_t stream) {
  const float* q  = (const float*)d_in[0];
  const float* k  = (const float*)d_in[1];
  const float* v  = (const float*)d_in[2];
  const float* Wq = (const float*)d_in[3];
  const float* bq = (const float*)d_in[4];
  const float* Wk = (const float*)d_in[5];
  const float* bk = (const float*)d_in[6];
  const float* Wv = (const float*)d_in[7];
  const float* bv = (const float*)d_in[8];
  const float* Wo = (const float*)d_in[9];
  const float* bo = (const float*)d_in[10];

  const size_t MD = 8192ull * 1024;  // activation elems
  const size_t DD = 1024ull * 1024;  // weight elems

  __bf16* ws  = (__bf16*)d_ws;
  __bf16* Xq  = ws;
  __bf16* Xk  = Xq + MD;
  __bf16* Xv  = Xk + MD;
  __bf16* Wqt = Xv + MD;    // [N][K] transposed bf16 weights
  __bf16* Wkt = Wqt + DD;
  __bf16* Wvt = Wkt + DD;
  __bf16* Wot = Wvt + DD;
  __bf16* Qh  = Wot + DD;   // [B,H,S,HD]
  __bf16* Kh  = Qh + MD;    // [B,H,S,HD]
  __bf16* Vt  = Kh + MD;    // [B,H,HD,S]
  __bf16* Ctx = Vt + MD;    // [B,S,H*HD]

  // activations f32 -> bf16
  cvt_f32_bf16<<<(int)(MD / 4 / 256), 256, 0, stream>>>(q, Xq, (int)(MD / 4));
  cvt_f32_bf16<<<(int)(MD / 4 / 256), 256, 0, stream>>>(k, Xk, (int)(MD / 4));
  cvt_f32_bf16<<<(int)(MD / 4 / 256), 256, 0, stream>>>(v, Xv, (int)(MD / 4));
  // weights f32 [K][N] -> bf16 [N][K] (once; GEMM tiles become pure b128 copies)
  dim3 tg(32, 32), tb(256);
  cvt_transpose_w<<<tg, tb, 0, stream>>>(Wq, Wqt);
  cvt_transpose_w<<<tg, tb, 0, stream>>>(Wk, Wkt);
  cvt_transpose_w<<<tg, tb, 0, stream>>>(Wv, Wvt);
  cvt_transpose_w<<<tg, tb, 0, stream>>>(Wo, Wot);

  // projections (M=8192, N=1024, K=1024), 128x128 blocks
  dim3 gg(64, 8), gb(256);
  gemm_bf16<0><<<gg, gb, 0, stream>>>(Xq, Wqt, bq, Qh, 8192, 1024, 1024);
  gemm_bf16<0><<<gg, gb, 0, stream>>>(Xk, Wkt, bk, Kh, 8192, 1024, 1024);
  gemm_bf16<1><<<gg, gb, 0, stream>>>(Xv, Wvt, bv, Vt, 8192, 1024, 1024);

  // attention: 128 q rows per block, (S/128, H, B)
  flash_attn<<<dim3(16, 16, 4), 256, 0, stream>>>(Qh, Kh, Vt, Ctx);

  // output projection -> f32
  gemm_bf16<2><<<gg, gb, 0, stream>>>(Ctx, Wot, bo, d_out, 8192, 1024, 1024);
}
